// SingularAttention_24026047054604
// MI455X (gfx1250) — compile-verified, matrix operand order checked by reading, not run on hardware
//
#include <hip/hip_runtime.h>
#include <hip/hip_bf16.h>

// ---------------------------------------------------------------------------
// SingularAttention on MI455X (gfx1250): bf16 WMMA pipeline, f32 accumulate.
// All WMMA K-loops use contiguous-in-K operands (global_load_b128 + v_wmma);
// transposed bf16 copies are produced once via LDS tile-transpose.
// ---------------------------------------------------------------------------

typedef __attribute__((ext_vector_type(16))) __bf16 v16bf;
typedef __attribute__((ext_vector_type(8)))  float  v8f;

#define B_   4
#define L_   4096
#define C_   1024
#define R_   256
#define M_   16
#define DK_  64
#define NEGV (-32767.0f)

union Frag { v16bf v; unsigned int u[8]; };

__device__ __forceinline__ unsigned int bf16pair(float lo, float hi) {
    unsigned int a = __float_as_uint(lo);
    unsigned int b = __float_as_uint(hi);
    a += 0x7fffu + ((a >> 16) & 1u);     // round-to-nearest-even
    b += 0x7fffu + ((b >> 16) & 1u);
    return (a >> 16) | (b & 0xffff0000u);
}
__device__ __forceinline__ unsigned short bf16one(float x) {
    unsigned int a = __float_as_uint(x);
    a += 0x7fffu + ((a >> 16) & 1u);
    return (unsigned short)(a >> 16);
}

// K index for packed pair j (0..7) given wave half (lane>>4):
// VGPR j<4: K = hf*8 + 2*(j&3); VGPR j>=4: +16.   (ISA 7.12.2, 16-bit A 16x32)
__device__ __forceinline__ int kidx(int j, int hf) {
    return ((j >> 2) << 4) + hf * 8 + ((j & 3) << 1);
}

// ---- bf16 fragment loaders (all contiguous-in-K -> b128-mergeable) --------
// A tile 16x32, row-major bf16 source
__device__ __forceinline__ v16bf ldA_bf(const unsigned short* a, int lda, int hf, int ln16) {
    Frag f; const unsigned short* row = a + (size_t)ln16 * lda;
#pragma unroll
    for (int j = 0; j < 8; ++j) { int kk = kidx(j, hf); f.u[j] = *(const unsigned int*)(row + kk); }
    return f.v;
}
// B tile 32x16, transposed source: B[k][n] = w[n*ldw + k] (k contiguous)
__device__ __forceinline__ v16bf ldBt_bf(const unsigned short* w, int ldw, int hf, int ln16) {
    Frag f; const unsigned short* row = w + (size_t)ln16 * ldw;
#pragma unroll
    for (int j = 0; j < 8; ++j) { int kk = kidx(j, hf); f.u[j] = *(const unsigned int*)(row + kk); }
    return f.v;
}

__device__ __forceinline__ v8f wmma_bf16(v16bf a, v16bf b, v8f c) {
    return __builtin_amdgcn_wmma_f32_16x16x32_bf16(false, a, false, b, (short)0, c, false, false);
}
__device__ __forceinline__ v8f zero8() { v8f z = {0.f,0.f,0.f,0.f,0.f,0.f,0.f,0.f}; return z; }

// ---------------------------------------------------------------------------
__global__ void k_zero(float* p) { if (threadIdx.x < 4) p[threadIdx.x] = 0.f; }

// plain f32 -> bf16 conversion (pair-packed u32 stores, grid-stride)
__global__ void k_cvt(const float* __restrict__ src, unsigned short* __restrict__ dst,
                      long long npairs) {
    long long i = (long long)blockIdx.x * blockDim.x + threadIdx.x;
    long long stride = (long long)gridDim.x * blockDim.x;
    unsigned int* d32 = (unsigned int*)dst;
    const float2* s2 = (const float2*)src;
    for (; i < npairs; i += stride) {
        float2 p = s2[i];
        d32[i] = bf16pair(p.x, p.y);
    }
}

// f32 [B][L][C] -> bf16 transposed [B][C][L] (and optionally row-major bf16 too)
// 32x32 tiles staged through padded LDS; all global accesses coalesced.
__global__ void k_cvtT(const float* __restrict__ src, unsigned short* __restrict__ dstT,
                       unsigned short* __restrict__ dstN) {
    __shared__ unsigned short lds[32][33];
    const int tilesC = C_ / 32;                        // 32
    const int tilesPerBatch = (L_ / 32) * tilesC;      // 4096
    int blk = blockIdx.x;
    int b = blk / tilesPerBatch, rem = blk % tilesPerBatch;
    int tl = rem / tilesC, tc = rem % tilesC;
    const float* sb = src + (size_t)b * L_ * C_;
    int tid = threadIdx.x;
    int rr = tid >> 5;          // 0..7
    int cc = tid & 31;          // 0..31
#pragma unroll
    for (int p = 0; p < 4; ++p) {
        int r = rr + p * 8;
        float x = sb[(size_t)(tl * 32 + r) * C_ + tc * 32 + cc];
        unsigned short h = bf16one(x);
        lds[cc][r] = h;
        if (dstN) dstN[(size_t)b * L_ * C_ + (size_t)(tl * 32 + r) * C_ + tc * 32 + cc] = h;
    }
    __syncthreads();
    unsigned int* d32 = (unsigned int*)(dstT + (size_t)b * C_ * L_);
#pragma unroll
    for (int p = 0; p < 2; ++p) {
        int w = tid + p * 256;                 // 0..511
        int c = w >> 4;                        // 0..31
        int lp = (w & 15) << 1;                // even l within tile
        unsigned int val = (unsigned int)lds[c][lp] | ((unsigned int)lds[c][lp + 1] << 16);
        d32[(((size_t)(tc * 32 + c)) * L_ + tl * 32 + lp) >> 1] = val;
    }
}

// pScore[B*L, R] = vx @ Wp^T + bp, masked   (all-bf16 operands, contiguous)
__global__ void k_pscore(const unsigned short* __restrict__ vxb, const unsigned short* __restrict__ wpb,
                         const float* __restrict__ bp, const unsigned char* __restrict__ maskPAD,
                         float* __restrict__ pScore) {
    int wid = (int)((blockIdx.x * blockDim.x + threadIdx.x) >> 5);
    int lane = threadIdx.x & 31, hf = lane >> 4, ln16 = lane & 15;
    const int Mt = (B_ * L_) / 16, Nt = R_ / 16;
    if (wid >= Mt * Nt) return;
    int mt = wid / Nt, nt = wid % Nt;
    const unsigned short* Ab = vxb + (size_t)mt * 16 * C_;
    const unsigned short* Wb = wpb + (size_t)(nt * 16) * C_;
    v8f acc = zero8();
    for (int k0 = 0; k0 < C_; k0 += 32) {
        v16bf a = ldA_bf(Ab + k0, C_, hf, ln16);
        v16bf b = ldBt_bf(Wb + k0, C_, hf, ln16);
        acc = wmma_bf16(a, b, acc);
    }
    int n = nt * 16 + ln16;
    float bias = bp[n];
#pragma unroll
    for (int i = 0; i < 8; ++i) {
        int row = mt * 16 + i + hf * 8;
        int bb = row >> 12, l = row & (L_ - 1);
        float val = acc[i] + bias;
        if (maskPAD[(size_t)bb * L_ * L_ + l] == 0) val = NEGV;
        pScore[(size_t)row * R_ + n] = val;
    }
}

// softmax over L (axis=-2): one block per (b,r); output bf16 [B,R,L]
__global__ void k_softmax_L(const float* __restrict__ pScore, unsigned short* __restrict__ pAlphaT) {
    int b = blockIdx.x / R_, r = blockIdx.x % R_;
    __shared__ float red[256];
    int tid = threadIdx.x;
    const float* col = pScore + (size_t)b * L_ * R_ + r;
    float m = -3.4e38f;
    for (int l = tid; l < L_; l += 256) m = fmaxf(m, col[(size_t)l * R_]);
    red[tid] = m; __syncthreads();
    for (int s = 128; s > 0; s >>= 1) { if (tid < s) red[tid] = fmaxf(red[tid], red[tid + s]); __syncthreads(); }
    m = red[0]; __syncthreads();
    float sum = 0.f;
    for (int l = tid; l < L_; l += 256) sum += __expf(col[(size_t)l * R_] - m);
    red[tid] = sum; __syncthreads();
    for (int s = 128; s > 0; s >>= 1) { if (tid < s) red[tid] += red[tid + s]; __syncthreads(); }
    float inv = 1.0f / red[0];
    unsigned short* outp = pAlphaT + (size_t)(b * R_ + r) * L_;
    for (int l = tid; l < L_; l += 256)
        outp[l] = bf16one(__expf(col[(size_t)l * R_] - m) * inv);
}

// softmax over R (axis=-1): one wave per (b,l); outputs bf16 [B,L,R] and [B,R,L]
__global__ void k_softmax_R(const float* __restrict__ pScore, unsigned short* __restrict__ pAlpha,
                            unsigned short* __restrict__ pAlphaT2) {
    int wid = (int)((blockIdx.x * blockDim.x + threadIdx.x) >> 5);
    int lane = threadIdx.x & 31;
    if (wid >= B_ * L_) return;
    int b = wid >> 12, l = wid & (L_ - 1);
    const float* row = pScore + (size_t)wid * R_;
    float v[8], m = -3.4e38f;
#pragma unroll
    for (int j = 0; j < 8; ++j) { v[j] = row[lane + 32 * j]; m = fmaxf(m, v[j]); }
#pragma unroll
    for (int off = 16; off > 0; off >>= 1) m = fmaxf(m, __shfl_xor(m, off, 32));
    float s = 0.f;
#pragma unroll
    for (int j = 0; j < 8; ++j) { v[j] = __expf(v[j] - m); s += v[j]; }
#pragma unroll
    for (int off = 16; off > 0; off >>= 1) s += __shfl_xor(s, off, 32);
    float inv = 1.f / s;
#pragma unroll
    for (int j = 0; j < 8; ++j) {
        int r = lane + 32 * j;
        unsigned short h = bf16one(v[j] * inv);
        pAlpha[(size_t)wid * R_ + r] = h;
        pAlphaT2[((size_t)b * R_ + r) * L_ + l] = h;
    }
}

// sum((P P^T * (1-I))^2) for P bf16 [B,R,L]; atomic into lossSlot
__global__ void k_gram(const unsigned short* __restrict__ P, float* lossSlot) {
    int wid = (int)((blockIdx.x * blockDim.x + threadIdx.x) >> 5);
    int lane = threadIdx.x & 31, hf = lane >> 4, ln16 = lane & 15;
    const int T = R_ / 16;
    if (wid >= B_ * T * T) return;
    int b = wid / (T * T), rem = wid % (T * T);
    int mt = rem / T, nt = rem % T;
    const unsigned short* Pb = P + (size_t)b * R_ * L_;
    const unsigned short* Ab = Pb + (size_t)(mt * 16) * L_;
    const unsigned short* Bb = Pb + (size_t)(nt * 16) * L_;
    v8f acc = zero8();
    for (int k0 = 0; k0 < L_; k0 += 32) {
        if (k0 + 128 < L_) {
            __builtin_prefetch(Ab + (size_t)ln16 * L_ + k0 + 128, 0, 1);
            __builtin_prefetch(Bb + (size_t)ln16 * L_ + k0 + 128, 0, 1);
        }
        v16bf a = ldA_bf(Ab + k0, L_, hf, ln16);
        v16bf bfr = ldBt_bf(Bb + k0, L_, hf, ln16);
        acc = wmma_bf16(a, bfr, acc);
    }
    float s = 0.f;
    int n = nt * 16 + ln16;
#pragma unroll
    for (int i = 0; i < 8; ++i) {
        int m = mt * 16 + i + hf * 8;
        float g = acc[i];
        if (m != n) s += g * g;
    }
#pragma unroll
    for (int off = 16; off > 0; off >>= 1) s += __shfl_xor(s, off, 32);
    if (lane == 0) atomicAdd(lossSlot, s);
}

// Out[b][R,C] = PT[b][R,L] @ X[b][L,C], with X given transposed XT[b][C,L]
__global__ void k_compress(const unsigned short* __restrict__ PT, const unsigned short* __restrict__ XT,
                           unsigned short* __restrict__ Out) {
    int wid = (int)((blockIdx.x * blockDim.x + threadIdx.x) >> 5);
    int lane = threadIdx.x & 31, hf = lane >> 4, ln16 = lane & 15;
    const int Mt = R_ / 16, Nt = C_ / 16, per = Mt * Nt;
    if (wid >= B_ * per) return;
    int b = wid / per, rem = wid % per;
    int mt = rem / Nt, nt = rem % Nt;
    const unsigned short* A  = PT + (size_t)b * R_ * L_ + (size_t)(mt * 16) * L_;
    const unsigned short* Bm = XT + (size_t)b * C_ * L_ + (size_t)(nt * 16) * L_;
    v8f acc = zero8();
    for (int k0 = 0; k0 < L_; k0 += 32) {
        if (k0 + 128 < L_) {
            __builtin_prefetch(A  + (size_t)ln16 * L_ + k0 + 128, 0, 1);
            __builtin_prefetch(Bm + (size_t)ln16 * L_ + k0 + 128, 0, 1);
        }
        v16bf a  = ldA_bf(A + k0, L_, hf, ln16);
        v16bf bf = ldBt_bf(Bm + k0, L_, hf, ln16);
        acc = wmma_bf16(a, bf, acc);
    }
    unsigned short* Ob = Out + (size_t)b * R_ * C_;
    int n = nt * 16 + ln16;
#pragma unroll
    for (int i = 0; i < 8; ++i)
        Ob[(size_t)(mt * 16 + i + hf * 8) * C_ + n] = bf16one(acc[i]);
}

// Out = A[b][R,Kdim] @ W^T + bias.
// mode 0: Out transposed [B][1024][R]   (packed b128 stores)
// mode 1: Out scatter    [B][M][R][DK]  (q,k head layout)
// mode 2: Out scatter-T  [B][M][DK][R]  (v head layout, packed b128 stores)
__global__ void k_wproj(const unsigned short* __restrict__ Ain, const unsigned short* __restrict__ W,
                        const float* __restrict__ bias, unsigned short* __restrict__ Out,
                        int Kdim, int mode) {
    int wid = (int)((blockIdx.x * blockDim.x + threadIdx.x) >> 5);
    int lane = threadIdx.x & 31, hf = lane >> 4, ln16 = lane & 15;
    const int Mt = R_ / 16, Nt = 1024 / 16, per = Mt * Nt;
    if (wid >= B_ * per) return;
    int b = wid / per, rem = wid % per;
    int mt = rem / Nt, nt = rem % Nt;
    const unsigned short* A  = Ain + (size_t)b * R_ * Kdim + (size_t)(mt * 16) * Kdim;
    const unsigned short* Wb = W + (size_t)(nt * 16) * Kdim;
    v8f acc = zero8();
    for (int k0 = 0; k0 < Kdim; k0 += 32) {
        v16bf a  = ldA_bf(A + k0, Kdim, hf, ln16);
        v16bf bf = ldBt_bf(Wb + k0, Kdim, hf, ln16);
        acc = wmma_bf16(a, bf, acc);
    }
    int n = nt * 16 + ln16;
    float bv = bias[n];
    int r0 = mt * 16 + hf * 8;          // 8 consecutive output rows per lane
    if (mode == 1) {
        int head = n >> 6, d = n & 63;
#pragma unroll
        for (int i = 0; i < 8; ++i)
            Out[(((size_t)(b * M_ + head)) * R_ + r0 + i) * DK_ + d] = bf16one(acc[i] + bv);
    } else {
        size_t base;
        if (mode == 2) { int head = n >> 6, d = n & 63; base = (((size_t)(b * M_ + head)) * DK_ + d) * R_ + r0; }
        else           { base = ((size_t)b * 1024 + n) * R_ + r0; }
        uint4 pk;
        pk.x = bf16pair(acc[0] + bv, acc[1] + bv);
        pk.y = bf16pair(acc[2] + bv, acc[3] + bv);
        pk.z = bf16pair(acc[4] + bv, acc[5] + bv);
        pk.w = bf16pair(acc[6] + bv, acc[7] + bv);
        *reinterpret_cast<uint4*>(Out + base) = pk;     // 16B aligned (r0 % 8 == 0)
    }
}

// fused scores (q k^T / 8) + row softmax over 256 cols + mean(alpha*omi) + store alpha bf16
__global__ void k_scores(const unsigned short* __restrict__ Q, const unsigned short* __restrict__ K,
                         unsigned short* __restrict__ AlphaOut, float* lossSlot) {
    int wid = (int)((blockIdx.x * blockDim.x + threadIdx.x) >> 5);
    int lane = threadIdx.x & 31, hf = lane >> 4, ln16 = lane & 15;
    const int strips = R_ / 16;
    if (wid >= B_ * M_ * strips) return;
    int bm = wid / strips, rt = wid % strips;
    const unsigned short* Qb = Q + (size_t)bm * R_ * DK_ + (size_t)(rt * 16) * DK_;
    const unsigned short* Kb = K + (size_t)bm * R_ * DK_;
    v8f acc[16];
#pragma unroll
    for (int t = 0; t < 16; ++t) acc[t] = zero8();
    v16bf a0 = ldA_bf(Qb,      DK_, hf, ln16);
    v16bf a1 = ldA_bf(Qb + 32, DK_, hf, ln16);
#pragma unroll
    for (int t = 0; t < 16; ++t) {
        const unsigned short* Kt = Kb + (size_t)(t * 16) * DK_;
        v16bf b0 = ldBt_bf(Kt,      DK_, hf, ln16);
        v16bf b1 = ldBt_bf(Kt + 32, DK_, hf, ln16);
        acc[t] = wmma_bf16(a0, b0, acc[t]);
        acc[t] = wmma_bf16(a1, b1, acc[t]);
    }
    float mx[8], sm[8];
#pragma unroll
    for (int i = 0; i < 8; ++i) { mx[i] = -3.4e38f; sm[i] = 0.f; }
#pragma unroll
    for (int t = 0; t < 16; ++t)
#pragma unroll
        for (int i = 0; i < 8; ++i) { float sv = acc[t][i] * 0.125f; acc[t][i] = sv; mx[i] = fmaxf(mx[i], sv); }
#pragma unroll
    for (int off = 1; off < 16; off <<= 1)
#pragma unroll
        for (int i = 0; i < 8; ++i) mx[i] = fmaxf(mx[i], __shfl_xor(mx[i], off, 32));
#pragma unroll
    for (int t = 0; t < 16; ++t)
#pragma unroll
        for (int i = 0; i < 8; ++i) { float e = __expf(acc[t][i] - mx[i]); acc[t][i] = e; sm[i] += e; }
#pragma unroll
    for (int off = 1; off < 16; off <<= 1)
#pragma unroll
        for (int i = 0; i < 8; ++i) sm[i] += __shfl_xor(sm[i], off, 32);
    float inv[8];
#pragma unroll
    for (int i = 0; i < 8; ++i) inv[i] = 1.f / sm[i];
    float lsum = 0.f;
    unsigned short* Ob = AlphaOut + (size_t)bm * R_ * R_;
#pragma unroll
    for (int t = 0; t < 16; ++t)
#pragma unroll
        for (int i = 0; i < 8; ++i) {
            float al = acc[t][i] * inv[i];
            int r = rt * 16 + i + hf * 8;
            int s = t * 16 + ln16;
            if (r != s) lsum += al;
            Ob[(size_t)r * R_ + s] = bf16one(al);
        }
#pragma unroll
    for (int off = 16; off > 0; off >>= 1) lsum += __shfl_xor(lsum, off, 32);
    if (lane == 0) atomicAdd(lossSlot, lsum);
}

// zcat[b][r][m*DK+d] = alpha[b,m] @ v[b,m] with V given transposed [B,M,DK,R]
__global__ void k_attnv(const unsigned short* __restrict__ Alpha, const unsigned short* __restrict__ VT,
                        unsigned short* __restrict__ Zcat) {
    int wid = (int)((blockIdx.x * blockDim.x + threadIdx.x) >> 5);
    int lane = threadIdx.x & 31, hf = lane >> 4, ln16 = lane & 15;
    const int Mt = R_ / 16, Nt = DK_ / 16, per = Mt * Nt;   // 16*4 = 64
    if (wid >= B_ * M_ * per) return;
    int bm = wid / per, rem = wid % per;
    int mt = rem / Nt, nt = rem % Nt;
    int b = bm / M_, m = bm % M_;
    const unsigned short* A  = Alpha + (size_t)bm * R_ * R_ + (size_t)(mt * 16) * R_;
    const unsigned short* Bm = VT + (size_t)bm * DK_ * R_ + (size_t)(nt * 16) * R_;
    v8f acc = zero8();
    for (int k0 = 0; k0 < R_; k0 += 32) {
        v16bf a  = ldA_bf(A + k0, R_, hf, ln16);
        v16bf bf = ldBt_bf(Bm + k0, R_, hf, ln16);
        acc = wmma_bf16(a, bf, acc);
    }
    unsigned short* Ob = Zcat + (size_t)b * R_ * (M_ * DK_);
    int n = nt * 16 + ln16;
#pragma unroll
    for (int i = 0; i < 8; ++i) {
        int r = mt * 16 + i + hf * 8;
        Ob[(size_t)r * (M_ * DK_) + m * DK_ + n] = bf16one(acc[i]);
    }
}

// out[b][L,C] = pAlpha_[b][L,R] @ zr[b][R,C] with zr given transposed [B][C][R]
__global__ void k_expand(const unsigned short* __restrict__ PA, const unsigned short* __restrict__ ZrT,
                         float* __restrict__ Out) {
    int wid = (int)((blockIdx.x * blockDim.x + threadIdx.x) >> 5);
    int lane = threadIdx.x & 31, hf = lane >> 4, ln16 = lane & 15;
    const int Mt = L_ / 16, Nt = C_ / 16, per = Mt * Nt;    // 256*64
    if (wid >= B_ * per) return;
    int b = wid / per, rem = wid % per;
    int mt = rem / Nt, nt = rem % Nt;
    const unsigned short* A  = PA + (size_t)b * L_ * R_ + (size_t)(mt * 16) * R_;
    const unsigned short* Bm = ZrT + (size_t)b * C_ * R_ + (size_t)(nt * 16) * R_;
    v8f acc = zero8();
    for (int k0 = 0; k0 < R_; k0 += 32) {
        v16bf a  = ldA_bf(A + k0, R_, hf, ln16);
        v16bf bf = ldBt_bf(Bm + k0, R_, hf, ln16);
        acc = wmma_bf16(a, bf, acc);
    }
    float* Ob = Out + (size_t)b * L_ * C_;
    int n = nt * 16 + ln16;
#pragma unroll
    for (int i = 0; i < 8; ++i)
        Ob[(size_t)(mt * 16 + i + hf * 8) * C_ + n] = acc[i];
}

__global__ void k_final(const float* __restrict__ addLossIn, const float* __restrict__ lossAcc,
                        float* __restrict__ outLoss) {
    if (threadIdx.x == 0 && blockIdx.x == 0) {
        float l1 = lossAcc[0] / (float)(B_ * R_ * R_);
        float l2 = lossAcc[1] / (float)((long long)B_ * M_ * R_ * R_);
        float l3 = lossAcc[2] / (float)(B_ * R_ * R_);
        *outLoss = addLossIn[0] + l1 + l2 + l3;
    }
}

// ---------------------------------------------------------------------------
extern "C" void kernel_launch(void* const* d_in, const int* in_sizes, int n_in,
                              void* d_out, int out_size, void* d_ws, size_t ws_size,
                              hipStream_t stream) {
    const float* qx = (const float*)d_in[0];
    const float* kx = (const float*)d_in[1];
    const float* vx = (const float*)d_in[2];
    const float* addLoss = (const float*)d_in[3];
    const unsigned char* maskPAD = (const unsigned char*)d_in[4];
    const float* Wp = (const float*)d_in[5];
    const float* bp = (const float*)d_in[6];
    const float* WQ = (const float*)d_in[7];
    const float* bQ = (const float*)d_in[8];
    const float* WK = (const float*)d_in[9];
    const float* bK = (const float*)d_in[10];
    const float* WV = (const float*)d_in[11];
    const float* bV = (const float*)d_in[12];
    const float* WO = (const float*)d_in[13];
    const float* bO = (const float*)d_in[14];
    float* out = (float*)d_out;

    char* w = (char*)d_ws;
    float* pScore = (float*)w;              w += (size_t)B_ * L_ * R_ * 4;
    unsigned short* pAlphaT  = (unsigned short*)w; w += (size_t)B_ * R_ * L_ * 2;  // [B,R,L]
    unsigned short* pAlpha2  = (unsigned short*)w; w += (size_t)B_ * L_ * R_ * 2;  // [B,L,R]
    unsigned short* pAlphaT2 = (unsigned short*)w; w += (size_t)B_ * R_ * L_ * 2;  // [B,R,L]
    unsigned short* qr = (unsigned short*)w; w += (size_t)B_ * R_ * C_ * 2;
    unsigned short* kr = (unsigned short*)w; w += (size_t)B_ * R_ * C_ * 2;
    unsigned short* vr = (unsigned short*)w; w += (size_t)B_ * R_ * C_ * 2;
    unsigned short* qh  = (unsigned short*)w; w += (size_t)B_ * M_ * R_ * DK_ * 2;  // [B,M,R,DK]
    unsigned short* kh  = (unsigned short*)w; w += (size_t)B_ * M_ * R_ * DK_ * 2;  // [B,M,R,DK]
    unsigned short* vhT = (unsigned short*)w; w += (size_t)B_ * M_ * DK_ * R_ * 2;  // [B,M,DK,R]
    unsigned short* alpha = (unsigned short*)w; w += (size_t)B_ * M_ * R_ * R_ * 2;
    unsigned short* zcat  = (unsigned short*)w; w += (size_t)B_ * R_ * M_ * DK_ * 2;
    unsigned short* zrT   = (unsigned short*)w; w += (size_t)B_ * C_ * R_ * 2;      // [B,C,R]
    unsigned short* qxT = (unsigned short*)w; w += (size_t)B_ * C_ * L_ * 2;        // [B,C,L]
    unsigned short* kxT = (unsigned short*)w; w += (size_t)B_ * C_ * L_ * 2;
    unsigned short* vxT = (unsigned short*)w; w += (size_t)B_ * C_ * L_ * 2;
    unsigned short* vxb = (unsigned short*)w; w += (size_t)B_ * L_ * C_ * 2;        // [B,L,C]
    unsigned short* wpb = (unsigned short*)w; w += (size_t)R_ * C_ * 2;
    unsigned short* wqb = (unsigned short*)w; w += (size_t)(M_ * DK_) * C_ * 2;
    unsigned short* wkb = (unsigned short*)w; w += (size_t)(M_ * DK_) * C_ * 2;
    unsigned short* wvb = (unsigned short*)w; w += (size_t)(M_ * DK_) * C_ * 2;
    unsigned short* wob = (unsigned short*)w; w += (size_t)C_ * (M_ * DK_) * 2;
    float* loss = (float*)w;                w += 16;

    const long long wPairs = (long long)(M_ * DK_) * C_ / 2;
    const int cvtTblocks = B_ * (L_ / 32) * (C_ / 32);   // 16384

    dim3 blk(256);
    k_zero<<<1, 32, 0, stream>>>(loss);
    // one-shot conversions / transposes (pure bandwidth passes)
    k_cvtT<<<cvtTblocks, blk, 0, stream>>>(qx, qxT, (unsigned short*)nullptr);
    k_cvtT<<<cvtTblocks, blk, 0, stream>>>(kx, kxT, (unsigned short*)nullptr);
    k_cvtT<<<cvtTblocks, blk, 0, stream>>>(vx, vxT, vxb);
    k_cvt<<<512, blk, 0, stream>>>(Wp, wpb, (long long)R_ * C_ / 2);
    k_cvt<<<512, blk, 0, stream>>>(WQ, wqb, wPairs);
    k_cvt<<<512, blk, 0, stream>>>(WK, wkb, wPairs);
    k_cvt<<<512, blk, 0, stream>>>(WV, wvb, wPairs);
    k_cvt<<<512, blk, 0, stream>>>(WO, wob, wPairs);

    k_pscore<<<2048, blk, 0, stream>>>(vxb, wpb, bp, maskPAD, pScore);
    k_softmax_L<<<B_ * R_, blk, 0, stream>>>(pScore, pAlphaT);
    k_softmax_R<<<2048, blk, 0, stream>>>(pScore, pAlpha2, pAlphaT2);
    k_gram<<<128, blk, 0, stream>>>(pAlphaT,  loss + 0);
    k_gram<<<128, blk, 0, stream>>>(pAlphaT2, loss + 2);
    k_compress<<<512, blk, 0, stream>>>(pAlphaT, qxT, qr);
    k_compress<<<512, blk, 0, stream>>>(pAlphaT, kxT, kr);
    k_compress<<<512, blk, 0, stream>>>(pAlphaT, vxT, vr);
    k_wproj<<<512, blk, 0, stream>>>(qr, wqb, bQ, qh,  C_, 1);
    k_wproj<<<512, blk, 0, stream>>>(kr, wkb, bK, kh,  C_, 1);
    k_wproj<<<512, blk, 0, stream>>>(vr, wvb, bV, vhT, C_, 2);
    k_scores<<<128, blk, 0, stream>>>(qh, kh, alpha, loss + 1);
    k_attnv<<<512, blk, 0, stream>>>(alpha, vhT, zcat);
    k_wproj<<<512, blk, 0, stream>>>(zcat, wob, bO, zrT, M_ * DK_, 0);
    k_expand<<<8192, blk, 0, stream>>>(pAlpha2, zrT, out);
    k_final<<<1, 32, 0, stream>>>(addLoss, loss, out + (size_t)B_ * L_ * C_);
}
